// Semantic_Component_Level_Memory_74345883894098
// MI455X (gfx1250) — matrix operand order repaired and checked by reading, not compiled, not run on hardware
//
#include <hip/hip_runtime.h>

// ---------------------------------------------------------------------------
// Problem constants
// ---------------------------------------------------------------------------
#define Hh   3
#define Kk   87
#define Ff   768
#define Bb   2048
#define Nn   43
#define HK   261          // H*K
#define HKP  272          // padded to 17 tiles of 16
#define ROWS 88064        // B*N  (== 5504 * 16 exactly)

// d_out layout (floats): f | S | k_flat
#define OUT_S_OFF     67633152L   // B*N*F
#define OUT_KFLAT_OFF 75294720L   // + B*N*K

// ws layout (bytes)
#define WS_KBF    0L              // 272*768 bf16            = 417792 B
#define WS_KNORM  417792L         // 272 f32
#define WS_HSTAR  418944L         // B*K int (stores hk row index)
#define WS_VAL    1131648L        // ROWS*272 f32  (~95.8 MB)

typedef __attribute__((ext_vector_type(16))) __bf16 v16bf;
typedef __attribute__((ext_vector_type(8)))  float  v8f;

union FragBF { uint4 q[2]; v16bf v; };

__device__ __forceinline__ unsigned short f2bf(float f) {
  __bf16 h = (__bf16)f;
  return __builtin_bit_cast(unsigned short, h);
}

// ---------------------------------------------------------------------------
// Kernel 1: k_param f32 -> bf16 (padded), row norms, and k_flat output
// grid = 272 blocks x 256
// ---------------------------------------------------------------------------
__global__ __launch_bounds__(256) void k_convert(const float* __restrict__ kp,
                                                 float* __restrict__ kflat,
                                                 unsigned short* __restrict__ kbf,
                                                 float* __restrict__ knorm) {
  __shared__ float red[256];
  int hk = blockIdx.x, t = threadIdx.x;
  float partial = 0.f;
  if (hk < HK) {
    const float* src = kp + (long)hk * Ff;
    int h = hk / Kk, k = hk - h * Kk;
    float* kf = kflat + ((long)k * Hh + h) * Ff;    // kt.reshape(-1,F) row = k*H+h
    for (int i = t; i < Ff; i += 256) {
      float v = src[i];
      partial += v * v;
      kbf[(long)hk * Ff + i] = f2bf(v);
      kf[i] = v;
    }
  } else {
    for (int i = t; i < Ff; i += 256) kbf[(long)hk * Ff + i] = 0;
  }
  red[t] = partial;
  __syncthreads();
  if (t == 0) {
    float s = 0.f;
    for (int i = 0; i < 256; ++i) s += red[i];
    knorm[hk] = s;
  }
}

// ---------------------------------------------------------------------------
// Kernel 2: distances via WMMA bf16 + per-row mean -> val = mean_h / d2
// grid = 5504 blocks x 128 (4 waves). Each block = 16 x-rows vs all 272 k-rows.
// ---------------------------------------------------------------------------
#define XS_STRIDE 1552   // bytes per LDS X row: 768 bf16 (1536B) + 16B pad
#define D2_STRIDE 276    // dwords per d2 row: 272 + 4 pad

__global__ __launch_bounds__(128) void k_dist(const float* __restrict__ x,
                                              const unsigned short* __restrict__ kbf,
                                              const float* __restrict__ knorm,
                                              float* __restrict__ val) {
  __shared__ __align__(16) unsigned char xs[16 * XS_STRIDE];
  __shared__ float d2buf[16 * D2_STRIDE];
  __shared__ float psum[128];
  __shared__ float xnorm[16];
  __shared__ float knl[HKP];
  __shared__ float meanl[16 * Hh];

  int t = threadIdx.x;
  long rowbase = (long)blockIdx.x * 16;

  for (int i = t; i < HKP; i += 128) knl[i] = knorm[i];

  // Load X tile (16 rows x 768 f32) -> bf16 LDS, deterministic norm partials.
  {
    int n = t >> 3, sub = t & 7;                // 8 threads per row
    const float4* src = (const float4*)(x + (rowbase + n) * Ff);
    float ss = 0.f;
#pragma unroll 4
    for (int j = 0; j < 24; ++j) {
      int c = sub + j * 8;                      // float4 index within row (192 total)
      float4 v = src[c];
      ss += v.x * v.x + v.y * v.y + v.z * v.z + v.w * v.w;
      uint2 p;
      p.x = (unsigned)f2bf(v.x) | ((unsigned)f2bf(v.y) << 16);
      p.y = (unsigned)f2bf(v.z) | ((unsigned)f2bf(v.w) << 16);
      *(uint2*)(xs + n * XS_STRIDE + c * 8) = p;
    }
    psum[t] = ss;
  }
  __syncthreads();
  if (t < 16) {
    float s = 0.f;
    for (int j = 0; j < 8; ++j) s += psum[t * 8 + j];
    xnorm[t] = s;
  }
  __syncthreads();

  // WMMA: A = kf tile (M=16 k-rows), B = X^T (N=16 x-rows), K = 768 in 24 steps
  int wave = t >> 5, lane = t & 31, lhi = lane >> 4, lm = lane & 15;
  const char* kbb = (const char*)kbf;
  for (int kt = wave; kt < 17; kt += 4) {
    v8f acc = {0.f, 0.f, 0.f, 0.f, 0.f, 0.f, 0.f, 0.f};
    const char* ab = kbb + (long)(kt * 16 + lm) * 1536 + lhi * 16;
    const unsigned char* bb = xs + lm * XS_STRIDE + lhi * 16;
#pragma unroll 4
    for (int fc = 0; fc < 24; ++fc) {
      FragBF a, bm;
      a.q[0]  = *(const uint4*)(ab + fc * 64);        // K = 8*lhi .. +7
      a.q[1]  = *(const uint4*)(ab + fc * 64 + 32);   // K = 16+8*lhi .. +7
      bm.q[0] = *(const uint4*)(bb + fc * 64);        // K = 8*lhi .. +7
      bm.q[1] = *(const uint4*)(bb + fc * 64 + 32);   // K = 16+8*lhi .. +7
      acc = __builtin_amdgcn_wmma_f32_16x16x32_bf16(false, a.v, false, bm.v,
                                                    (short)0, acc, false, false);
    }
    float xn = xnorm[lm];
#pragma unroll
    for (int q = 0; q < 8; ++q) {
      int krow = kt * 16 + lhi * 8 + q;
      float d2 = knl[krow] + xn - 2.f * acc[q];
      d2buf[lm * D2_STRIDE + krow] = fmaxf(d2, 0.f);
    }
  }
  __syncthreads();

  // mean over K per (row, h)
  if (t < 16 * Hh) {
    int n = t / Hh, h = t - n * Hh;
    const float* dp = &d2buf[n * D2_STRIDE + h * Kk];
    float s = 0.f;
    for (int k = 0; k < Kk; ++k) s += dp[k];
    meanl[t] = s * (1.f / (float)Kk);
  }
  __syncthreads();

  // val[b,n,h,k] = mean / d2  (coalesced, stride-272 rows)
  float* vout = val + rowbase * HKP;
  for (int i = t; i < 16 * HKP; i += 128) {
    int n = i / HKP, col = i - n * HKP;
    float v = 0.f;
    if (col < HK) v = meanl[n * Hh + col / Kk] / d2buf[n * D2_STRIDE + col];
    vout[i] = v;
  }
}

// ---------------------------------------------------------------------------
// Kernel 3: per-b softmax S + h_star (no global atomics).  grid = 2048 x 256
// ---------------------------------------------------------------------------
#define VB_STRIDE 264

__global__ __launch_bounds__(256) void k_softmax(const float* __restrict__ val,
                                                 const unsigned char* __restrict__ mask,
                                                 float* __restrict__ Sout,
                                                 int* __restrict__ hstar) {
  __shared__ float vb[Nn * VB_STRIDE];
  __shared__ float sb[Nn * 88];
  __shared__ float rmax[Nn], rinv[Nn];
  int b = blockIdx.x, t = threadIdx.x;

  const float* vsrc = val + (long)b * Nn * HKP;
  for (int i = t; i < Nn * HKP; i += 256) {
    int n = i / HKP, col = i - n * HKP;
    if (col < HK) vb[n * VB_STRIDE + col] = vsrc[i];
  }
  __syncthreads();

  for (int i = t; i < Nn * 88; i += 256) {
    int n = i / 88, k = i - n * 88;
    if (k < Kk) {
      float v0 = vb[n * VB_STRIDE + k];
      float v1 = vb[n * VB_STRIDE + Kk + k];
      float v2 = vb[n * VB_STRIDE + 2 * Kk + k];
      float m = fmaxf(v0, fmaxf(v1, v2));
      float mk = mask[(long)b * Nn + n] ? 1.f : 0.f;
      sb[n * 88 + k] = m * 5.f * mk;
    }
  }
  __syncthreads();

  if (t < Nn) {
    float m = -3.4e38f;
    for (int k = 0; k < Kk; ++k) m = fmaxf(m, sb[t * 88 + k]);
    float s = 0.f;
    for (int k = 0; k < Kk; ++k) s += expf(sb[t * 88 + k] - m);
    rmax[t] = m;
    rinv[t] = 1.f / s;
  }
  __syncthreads();

  float* So = Sout + (long)b * Nn * Kk;
  for (int i = t; i < Nn * Kk; i += 256) {
    int n = i / Kk, k = i - n * Kk;
    So[i] = expf(sb[n * 88 + k] - rmax[n]) * rinv[n];
  }

  if (t < Kk) {
    int besth = 0;
    float bv = -3.4e38f;
    for (int h = 0; h < Hh; ++h) {
      float m = -3.4e38f;
      for (int n = 0; n < Nn; ++n) m = fmaxf(m, vb[n * VB_STRIDE + h * Kk + t]);
      if (m > bv) { bv = m; besth = h; }    // first-max tie-break like argmax
    }
    hstar[(long)b * Kk + t] = besth * Kk + t;   // store kbf row index directly
  }
}

// ---------------------------------------------------------------------------
// Kernel 4: f[b] = S[b] @ gather(kbf, hstar[b]) via WMMA bf16.  grid = 2048 x 256
// ---------------------------------------------------------------------------
#define SBF_STRIDE 208   // 96 bf16 (192B) + 16B pad
#define GR_STRIDE  272   // 128 bf16 (256B) + 16B pad
#define GT_STRIDE  208   // 96 bf16 + pad

__global__ __launch_bounds__(256) void k_einsum(const float* __restrict__ Sout,
                                                const int* __restrict__ hstar,
                                                const unsigned short* __restrict__ kbf,
                                                float* __restrict__ fout) {
  __shared__ int hs[Kk];
  __shared__ __align__(16) unsigned char sbf[48 * SBF_STRIDE];   // A: S, 48x96 bf16
  __shared__ __align__(16) unsigned char grows[96 * GR_STRIDE];  // gathered rows [k][f-chunk]
  __shared__ __align__(16) unsigned char gt[128 * GT_STRIDE];    // transposed [f][k]
  int b = blockIdx.x, t = threadIdx.x;

  if (t < Kk) hs[t] = hstar[(long)b * Kk + t];

  // A tile: S (43x87 f32) -> bf16, padded to 48x96
  const float* Ss = Sout + (long)b * Nn * Kk;
  for (int i = t; i < 48 * 96; i += 256) {
    int n = i / 96, k = i - n * 96;
    float v = (n < Nn && k < Kk) ? Ss[n * Kk + k] : 0.f;
    *(unsigned short*)(sbf + n * SBF_STRIDE + k * 2) = f2bf(v);
  }
  __syncthreads();

  int wave = t >> 5, lane = t & 31, lhi = lane >> 4, lm = lane & 15;
  float* outb = fout + (long)b * Nn * Ff;

  for (int fc = 0; fc < 6; ++fc) {              // 6 chunks of 128 features
    // gather 87 selected k rows (this f-chunk) into LDS, zero-pad to 96
    for (int i = t; i < 96 * 16; i += 256) {
      int r = i >> 4, c = i & 15;
      uint4 v;
      if (r < Kk) {
        const uint4* src = (const uint4*)((const char*)kbf + (long)hs[r] * 1536 + fc * 256);
        v = src[c];
      } else {
        v.x = v.y = v.z = v.w = 0u;
      }
      *(uint4*)(grows + r * GR_STRIDE + c * 16) = v;
    }
    __syncthreads();
    // LDS transpose: [k][f] -> [f][k] (pair k's into b32 writes)
    for (int i = t; i < 48 * 128; i += 256) {
      int kp = i >> 7, f = i & 127;
      unsigned a0 = *(const unsigned short*)(grows + (2 * kp)     * GR_STRIDE + f * 2);
      unsigned a1 = *(const unsigned short*)(grows + (2 * kp + 1) * GR_STRIDE + f * 2);
      *(unsigned*)(gt + f * GT_STRIDE + kp * 4) = a0 | (a1 << 16);
    }
    __syncthreads();

    // 3 M-tiles x 8 N-tiles in this chunk; K = 96 in 3 WMMA steps
    for (int tl = wave; tl < 24; tl += 8) {
      int mt = tl >> 3, ntl = tl & 7;
      v8f acc = {0.f, 0.f, 0.f, 0.f, 0.f, 0.f, 0.f, 0.f};
#pragma unroll
      for (int c = 0; c < 3; ++c) {
        FragBF a, bm;
        const unsigned char* ab = sbf + (mt * 16 + lm) * SBF_STRIDE + c * 64 + lhi * 16;
        a.q[0] = *(const uint4*)(ab);
        a.q[1] = *(const uint4*)(ab + 32);
        const unsigned char* bb = gt + (ntl * 16 + lm) * GT_STRIDE + c * 64 + lhi * 16;
        bm.q[0] = *(const uint4*)(bb);
        bm.q[1] = *(const uint4*)(bb + 32);
        acc = __builtin_amdgcn_wmma_f32_16x16x32_bf16(false, a.v, false, bm.v,
                                                      (short)0, acc, false, false);
      }
      int fcol = fc * 128 + ntl * 16 + lm;
#pragma unroll
      for (int q = 0; q < 8; ++q) {
        int m = mt * 16 + lhi * 8 + q;
        if (m < Nn) outb[(long)m * Ff + fcol] = acc[q];
      }
    }
    __syncthreads();
  }
}

// ---------------------------------------------------------------------------
extern "C" void kernel_launch(void* const* d_in, const int* in_sizes, int n_in,
                              void* d_out, int out_size, void* d_ws, size_t ws_size,
                              hipStream_t stream) {
  (void)in_sizes; (void)n_in; (void)out_size; (void)ws_size;
  const float* x            = (const float*)d_in[0];
  const unsigned char* mask = (const unsigned char*)d_in[1];
  const float* kp           = (const float*)d_in[2];

  float* out        = (float*)d_out;
  float* f_out      = out;
  float* S_out      = out + OUT_S_OFF;
  float* kflat_out  = out + OUT_KFLAT_OFF;

  unsigned char* ws = (unsigned char*)d_ws;
  unsigned short* kbf = (unsigned short*)(ws + WS_KBF);
  float* knorm        = (float*)(ws + WS_KNORM);
  int*   hstar        = (int*)(ws + WS_HSTAR);
  float* val          = (float*)(ws + WS_VAL);

  k_convert<<<HKP, 256, 0, stream>>>(kp, kflat_out, kbf, knorm);
  k_dist<<<ROWS / 16, 128, 0, stream>>>(x, kbf, knorm, val);
  k_softmax<<<Bb, 256, 0, stream>>>(val, mask, S_out, hstar);
  k_einsum<<<Bb, 256, 0, stream>>>(S_out, hstar, kbf, f_out);
}
